// CMOEBlock_26766236189237
// MI455X (gfx1250) — matrix operand order repaired and checked by reading, not compiled, not run on hardware
//
#include <hip/hip_runtime.h>
#include <hip/hip_bf16.h>

typedef _Float16 half_t;
typedef __attribute__((ext_vector_type(16))) _Float16 v16h;
typedef __attribute__((ext_vector_type(8)))  _Float16 v8h;
typedef __attribute__((ext_vector_type(8)))  float    v8f;
typedef int v4i_vs __attribute__((vector_size(16)));   // matches builtin param type
typedef __attribute__((address_space(1))) v4i_vs* g4p; // global int4*
typedef __attribute__((address_space(3))) v4i_vs* l4p; // LDS int4*

#define NPATCH 72          // 2 * 6 * 6
#define PPX    4096        // 64*64 pixels per patch
#define HW     147456      // 384*384
#define NPIX   294912      // 2*384*384

#if defined(__has_builtin)
#if __has_builtin(__builtin_amdgcn_global_load_async_to_lds_b128) && \
    __has_builtin(__builtin_amdgcn_s_wait_asynccnt)
#define HAVE_ASYNC_LDS 1
#else
#define HAVE_ASYNC_LDS 0
#endif
#else
#define HAVE_ASYNC_LDS 0
#endif

// ---------------------------------------------------------------------------
// WMMA helpers (CDNA5 wave32, f16 operands / f32 accumulate)
// A 16x32 f16: lane r=lane&15 is row M=r; v16h = { row[kc+hi*8 .. +7], row[kc+16+hi*8 .. +7] }
// B 32x16 f16 (loaded from B^T stored [N][K]): lane col = lane&15; v16h = { row[kc+hi*8 .. +7], row[kc+16+hi*8 .. +7] }
// D f32 16x16: lane (col = lane&15) holds 8 consecutive M values starting at hi*8
// ---------------------------------------------------------------------------
__device__ __forceinline__ v16h load_a_tile(const half_t* __restrict__ wtile, int ldk, int kc) {
    const int lane = threadIdx.x & 31;
    const int r = lane & 15, hi = lane >> 4;
    const half_t* p = wtile + (size_t)r * ldk + kc + hi * 8;
    union { v16h v; v8h h[2]; } u;
    u.h[0] = *(const v8h*)(p);
    u.h[1] = *(const v8h*)(p + 16);
    return u.v;
}

__device__ __forceinline__ v16h load_b_row(const half_t* rowp, int kc) {
    const int hi = (threadIdx.x >> 4) & 1;
    const half_t* p = rowp + kc + hi * 8;
    union { v16h v; v8h h[2]; } u;
    u.h[0] = *(const v8h*)(p);
    u.h[1] = *(const v8h*)(p + 16);
    return u.v;
}

__device__ __forceinline__ v8f wmma_f16(v16h a, v16h b, v8f c) {
    return __builtin_amdgcn_wmma_f32_16x16x32_f16(false, a, false, b, (short)0, c, false, false);
}

// ---------------------------------------------------------------------------
// Per-patch expert weight aggregation: aw[n] = sum_k v[n,k] * w[k]  (f16 out)
// ---------------------------------------------------------------------------
__global__ void agg_pw_kernel(const float* __restrict__ w, const float* __restrict__ b,
                              const float* __restrict__ v, half_t* __restrict__ awh,
                              float* __restrict__ ab, int M, int KC) {
    int id = blockIdx.x * 256 + threadIdx.x;
    const int wtot = NPATCH * M * KC;
    if (id < wtot) {
        int n = id / (M * KC);
        int r = id - n * (M * KC);
        int bb = n / 36, pp = n - bb * 36;
        float s = 0.f;
        #pragma unroll
        for (int k = 0; k < 5; ++k)
            s += v[(bb * 5 + k) * 36 + pp] * w[(size_t)k * M * KC + r];
        awh[(size_t)n * M * KC + r] = (half_t)s;
    } else if (id < wtot + NPATCH * M) {
        int id2 = id - wtot;
        int n = id2 / M, o = id2 - n * M;
        int bb = n / 36, pp = n - bb * 36;
        float s = 0.f;
        #pragma unroll
        for (int k = 0; k < 5; ++k)
            s += v[(bb * 5 + k) * 36 + pp] * b[k * M + o];
        ab[n * M + o] = s;
    }
}

// 3x3 weights: w2 [5][128][128][3][3] -> aw2h [72][tap=dy*3+dx][128][128] f16
__global__ void agg_w2_kernel(const float* __restrict__ w, const float* __restrict__ b,
                              const float* __restrict__ v, half_t* __restrict__ awh,
                              float* __restrict__ ab) {
    int id = blockIdx.x * 256 + threadIdx.x;
    const int wtot = NPATCH * 9 * 128 * 128;
    if (id < wtot) {
        int n = id / (9 * 128 * 128);
        int r = id - n * (9 * 128 * 128);
        int tap = r / (128 * 128);
        int rr = r - tap * (128 * 128);
        int o = rr >> 7, c = rr & 127;
        int dy = tap / 3, dx = tap - dy * 3;
        int bb = n / 36, pp = n - bb * 36;
        float s = 0.f;
        #pragma unroll
        for (int k = 0; k < 5; ++k)
            s += v[(bb * 5 + k) * 36 + pp] *
                 w[((((size_t)k * 128 + o) * 128 + c) * 3 + dy) * 3 + dx];
        awh[(size_t)n * 9 * 128 * 128 + r] = (half_t)s;
    } else if (id < wtot + NPATCH * 128) {
        int id2 = id - wtot;
        int n = id2 >> 7, o = id2 & 127;
        int bb = n / 36, pp = n - bb * 36;
        float s = 0.f;
        #pragma unroll
        for (int k = 0; k < 5; ++k)
            s += v[(bb * 5 + k) * 36 + pp] * b[k * 128 + o];
        ab[n * 128 + o] = s;
    }
}

__global__ void zero_msum_kernel(float* __restrict__ msum) {
    if (threadIdx.x < 128) msum[threadIdx.x] = 0.f;
}

// ---------------------------------------------------------------------------
// LN1: NCHW f32 in -> per-pixel channel LayerNorm -> patch layout [n][pix][64] f16
// ---------------------------------------------------------------------------
__global__ void ln1_kernel(const float* __restrict__ inp, const float* __restrict__ g,
                           const float* __restrict__ be, half_t* __restrict__ out) {
    int id = blockIdx.x * 256 + threadIdx.x;
    if (id >= NPIX) return;
    int b = id / HW;
    int r = id - b * HW;
    int Y = r / 384, X = r - Y * 384;
    const float* base = inp + ((size_t)b * 64 * 384 + Y) * 384 + X;
    float s = 0.f, s2 = 0.f;
    for (int c = 0; c < 64; ++c) {
        float x = base[(size_t)c * HW];
        s += x; s2 += x * x;
    }
    float mu  = s * (1.f / 64.f);
    float var = s2 * (1.f / 64.f) - mu * mu;
    float inv = rsqrtf(var + 1e-6f);
    int n = b * 36 + (Y >> 6) * 6 + (X >> 6);
    int p = ((Y & 63) << 6) | (X & 63);
    half_t* o = out + ((size_t)n * PPX + p) * 64;
    for (int c = 0; c < 64; ++c) {
        float x = base[(size_t)c * HW];
        o[c] = (half_t)(g[c] * (x - mu) * inv + be[c]);
    }
}

// LN2: patch layout f32 (contiguous channels) -> patch layout f16
__global__ void ln2_kernel(const float* __restrict__ y, const float* __restrict__ g,
                           const float* __restrict__ be, half_t* __restrict__ out) {
    int id = blockIdx.x * 256 + threadIdx.x;
    if (id >= NPATCH * PPX) return;
    const float* p = y + (size_t)id * 64;
    float s = 0.f, s2 = 0.f;
    for (int c = 0; c < 64; ++c) { float x = p[c]; s += x; s2 += x * x; }
    float mu  = s * (1.f / 64.f);
    float var = s2 * (1.f / 64.f) - mu * mu;
    float inv = rsqrtf(var + 1e-6f);
    half_t* o = out + (size_t)id * 64;
    for (int c = 0; c < 64; ++c)
        o[c] = (half_t)(g[c] * (p[c] - mu) * inv + be[c]);
}

// ---------------------------------------------------------------------------
// SimpleGate: x[.,c]*x[.,c+64] (128->64 ch). Optionally accumulate global mean.
// ---------------------------------------------------------------------------
__global__ void gate_kernel(const half_t* __restrict__ x, half_t* __restrict__ out,
                            float* __restrict__ msum, int do_mean) {
    __shared__ float ssum[64];
    const int tid = threadIdx.x;
    const int c = tid & 63, sub = tid >> 6;
    if (do_mean) {
        if (tid < 64) ssum[tid] = 0.f;
        __syncthreads();
    }
    const int pbase = blockIdx.x * 64;
    float acc = 0.f;
    #pragma unroll
    for (int i = 0; i < 16; ++i) {
        int p = pbase + sub + 4 * i;
        float a = (float)x[(size_t)p * 128 + c];
        float bq = (float)x[(size_t)p * 128 + 64 + c];
        float k = a * bq;
        out[(size_t)p * 64 + c] = (half_t)k;
        acc += k;
    }
    if (do_mean) {
        atomicAdd(&ssum[c], acc);
        __syncthreads();
        if (tid < 64) {
            int bI = pbase / HW;
            atomicAdd(&msum[bI * 64 + tid], ssum[tid]);
        }
    }
}

// ---------------------------------------------------------------------------
// Generic per-patch pointwise GEMM: out = W(MxKC) * act^T + bias, WMMA f16.
// EPI 0: store f16; EPI 1: *global mean, f16; EPI 2: inp + o*beta -> f32 patch;
// EPI 3: ybuf + o*gamma -> f32 NCHW (d_out)
// ---------------------------------------------------------------------------
template <int M, int KC, int EPI>
__global__ __launch_bounds__(256)
void gemm_pw_kernel(const half_t* __restrict__ W, const float* __restrict__ bias,
                    const half_t* __restrict__ act, half_t* __restrict__ outh,
                    float* __restrict__ outf, const float* __restrict__ msum,
                    const float* __restrict__ resid, const float* __restrict__ scale) {
    const int wid = (blockIdx.x * 256 + threadIdx.x) >> 5;
    const int tilesPerN = (M / 16) * 64;
    const int n = wid / tilesPerN;
    if (n >= NPATCH) return;
    int rem = wid - n * tilesPerN;
    const int mt = rem >> 6;
    const int pg = rem & 63;

    const half_t* Wt = W + ((size_t)n * M + mt * 16) * KC;
    const half_t* An = act + ((size_t)n * PPX + (size_t)pg * 64) * KC;
    const int lane = threadIdx.x & 31;
    const int col = lane & 15, hi = lane >> 4;

    v8f acc[4] = {};
    for (int kc = 0; kc < KC; kc += 32) {
        v16h a = load_a_tile(Wt, KC, kc);
        #pragma unroll
        for (int j = 0; j < 4; ++j) {
            v16h b = load_b_row(An + (size_t)(j * 16 + col) * KC, kc);
            acc[j] = wmma_f16(a, b, acc[j]);
        }
    }

    const int mbase = mt * 16 + hi * 8;
    float bs[8];
    #pragma unroll
    for (int i = 0; i < 8; ++i) bs[i] = bias[n * M + mbase + i];

    float mf[8];
    if (EPI == 1) {
        #pragma unroll
        for (int i = 0; i < 8; ++i)
            mf[i] = msum[(n / 36) * 64 + mbase + i] * (1.f / (float)HW);
    }

    #pragma unroll
    for (int j = 0; j < 4; ++j) {
        const int p = pg * 64 + j * 16 + col;
        if (EPI == 0) {
            v8h o;
            #pragma unroll
            for (int i = 0; i < 8; ++i) o[i] = (half_t)(acc[j][i] + bs[i]);
            *(v8h*)(outh + ((size_t)n * PPX + p) * M + mbase) = o;
        } else if (EPI == 1) {
            v8h o;
            #pragma unroll
            for (int i = 0; i < 8; ++i) o[i] = (half_t)(mf[i] * (acc[j][i] + bs[i]));
            *(v8h*)(outh + ((size_t)n * PPX + p) * M + mbase) = o;
        } else if (EPI == 2) {
            int bI = n / 36, pr = n - bI * 36;
            int Y = (pr / 6) * 64 + (p >> 6), X = (pr % 6) * 64 + (p & 63);
            #pragma unroll
            for (int i = 0; i < 8; ++i) {
                int c = mbase + i;
                float r = resid[(((size_t)bI * 64 + c) * 384 + Y) * 384 + X];
                outf[((size_t)n * PPX + p) * 64 + c] = r + (acc[j][i] + bs[i]) * scale[c];
            }
        } else {
            int bI = n / 36, pr = n - bI * 36;
            int Y = (pr / 6) * 64 + (p >> 6), X = (pr % 6) * 64 + (p & 63);
            #pragma unroll
            for (int i = 0; i < 8; ++i) {
                int c = mbase + i;
                float yv = resid[((size_t)n * PPX + p) * 64 + c];
                outf[(((size_t)bI * 64 + c) * 384 + Y) * 384 + X] =
                    yv + (acc[j][i] + bs[i]) * scale[c];
            }
        }
    }
}

// ---------------------------------------------------------------------------
// 3x3 per-patch conv (128->128), workgroup-cooperative:
// block = one (patch n, row y); 8 waves = 8 M-tiles; 3 activation rows staged
// into LDS with zero halo columns/rows (async global->LDS when available),
// then 9 shifted WMMA GEMMs with unconditional ds_load reads (no exec masks).
// LDS pixel stride padded to 136 halves (68 dwords) to kill bank conflicts.
// ---------------------------------------------------------------------------
#define LPS 136                      // halves per pixel slot in LDS (pad 128->136)
#define LROW (68 * LPS)              // halves per staged row (66 used + pad)

__global__ __launch_bounds__(256)
void conv3x3_kernel(const half_t* __restrict__ W9, const float* __restrict__ bias,
                    const half_t* __restrict__ act, half_t* __restrict__ out) {
    __shared__ __align__(16) half_t smem[3 * LROW];   // 55,488 B

    const int n = blockIdx.x / 64;
    const int y = blockIdx.x - n * 64;
    const int tid = threadIdx.x;
    const half_t* An = act + (size_t)n * PPX * 128;

    // ---- stage 3 rows (y-1, y, y+1), x in [0,64), 128 ch f16, b128 chunks ----
    const v8h zero8 = {};
    // halo columns (slot 0 => x=-1, slot 65 => x=64): 3 rows * 2 slots * 16 chunks
    if (tid < 96) {
        int r = tid >> 5, rem = tid & 31;
        int slot = (rem >> 4) ? 65 : 0;
        int k8 = (rem & 15) * 8;
        *(v8h*)(smem + (r * 68 + slot) * LPS + k8) = zero8;
    }
    // main data: 3 rows * 64 px * 16 chunks = 3072 chunks, 12 per thread
    for (int cid = tid; cid < 3072; cid += 256) {
        int r = cid >> 10;                 // 0..2
        int rem = cid & 1023;
        int px = rem >> 4;                 // 0..63
        int k8 = (rem & 15) * 8;           // 0..120
        int yy = y - 1 + r;
        half_t* dst = smem + (r * 68 + 1 + px) * LPS + k8;
        if ((unsigned)yy < 64u) {
            const half_t* src = An + ((size_t)(yy * 64 + px)) * 128 + k8;
#if HAVE_ASYNC_LDS
            __builtin_amdgcn_global_load_async_to_lds_b128((g4p)src, (l4p)dst, 0, 0);
#else
            *(v8h*)dst = *(const v8h*)src;
#endif
        } else {
            *(v8h*)dst = zero8;            // zero row at patch border
        }
    }
#if HAVE_ASYNC_LDS
    __builtin_amdgcn_s_wait_asynccnt(0);
#endif
    __syncthreads();

    // ---- compute: wave = M-tile; 4 pixel tiles x 9 taps x 4 k-chunks ----
    const int mt = tid >> 5;
    const int lane = tid & 31;
    const int col = lane & 15, hi = lane >> 4;

    v8f acc[4] = {};
    for (int tap = 0; tap < 9; ++tap) {
        const int dy = tap / 3, dx = tap % 3;      // 0..2 each
        const half_t* Wt = W9 + (((size_t)n * 9 + tap) * 128 + mt * 16) * 128;
        const half_t* lrow = smem + dy * LROW;
        #pragma unroll
        for (int kc = 0; kc < 128; kc += 32) {
            v16h a = load_a_tile(Wt, 128, kc);
            #pragma unroll
            for (int j = 0; j < 4; ++j) {
                // slot = 1 + x + (dx-1); x = j*16+col  ->  slot = j*16+col+dx
                v16h b = load_b_row(lrow + (j * 16 + col + dx) * LPS, kc);
                acc[j] = wmma_f16(a, b, acc[j]);
            }
        }
    }

    const int mbase = mt * 16 + hi * 8;
    float bs[8];
    #pragma unroll
    for (int i = 0; i < 8; ++i) bs[i] = bias[n * 128 + mbase + i];
    #pragma unroll
    for (int j = 0; j < 4; ++j) {
        const int p = y * 64 + j * 16 + col;
        v8h o;
        #pragma unroll
        for (int i = 0; i < 8; ++i) o[i] = (half_t)(acc[j][i] + bs[i]);
        *(v8h*)(out + ((size_t)n * PPX + p) * 128 + mbase) = o;
    }
}

// ---------------------------------------------------------------------------
extern "C" void kernel_launch(void* const* d_in, const int* in_sizes, int n_in,
                              void* d_out, int out_size, void* d_ws, size_t ws_size,
                              hipStream_t stream) {
    const float* inp  = (const float*)d_in[0];
    const float* v    = (const float*)d_in[1];
    const float* w1   = (const float*)d_in[2];
    const float* b1   = (const float*)d_in[3];
    const float* w2   = (const float*)d_in[4];
    const float* b2   = (const float*)d_in[5];
    const float* w3   = (const float*)d_in[6];
    const float* b3   = (const float*)d_in[7];
    const float* wsca = (const float*)d_in[8];
    const float* bsca = (const float*)d_in[9];
    const float* w4   = (const float*)d_in[10];
    const float* b4   = (const float*)d_in[11];
    const float* w5   = (const float*)d_in[12];
    const float* b5   = (const float*)d_in[13];
    const float* n1w  = (const float*)d_in[14];
    const float* n1b  = (const float*)d_in[15];
    const float* n2w  = (const float*)d_in[16];
    const float* n2b  = (const float*)d_in[17];
    const float* beta = (const float*)d_in[18];
    const float* gam  = (const float*)d_in[19];
    float* dout = (float*)d_out;

    char* ws = (char*)d_ws;
    size_t off = 0;
    auto alloc = [&](size_t bytes) -> void* {
        void* p = ws + off;
        off = (off + bytes + 255) & ~(size_t)255;
        return p;
    };

    half_t* aw1h = (half_t*)alloc((size_t)NPATCH * 128 * 64 * 2);
    float*  ab1  = (float*) alloc((size_t)NPATCH * 128 * 4);
    half_t* aw2h = (half_t*)alloc((size_t)NPATCH * 9 * 128 * 128 * 2);
    float*  ab2  = (float*) alloc((size_t)NPATCH * 128 * 4);
    half_t* awsh = (half_t*)alloc((size_t)NPATCH * 64 * 64 * 2);
    float*  absc = (float*) alloc((size_t)NPATCH * 64 * 4);
    half_t* aw3h = (half_t*)alloc((size_t)NPATCH * 64 * 64 * 2);
    float*  ab3  = (float*) alloc((size_t)NPATCH * 64 * 4);
    half_t* aw4h = (half_t*)alloc((size_t)NPATCH * 128 * 64 * 2);
    float*  ab4  = (float*) alloc((size_t)NPATCH * 128 * 4);
    half_t* aw5h = (half_t*)alloc((size_t)NPATCH * 64 * 64 * 2);
    float*  ab5  = (float*) alloc((size_t)NPATCH * 64 * 4);
    float*  msum = (float*) alloc(128 * 4);
    half_t* bufA = (half_t*)alloc((size_t)NPATCH * PPX * 128 * 2); // x1 / z
    half_t* bufB = (half_t*)alloc((size_t)NPATCH * PPX * 128 * 2); // x2
    half_t* bufC = (half_t*)alloc((size_t)NPATCH * PPX * 64 * 2);  // xln / keep / yln
    half_t* bufD = (half_t*)alloc((size_t)NPATCH * PPX * 64 * 2);  // t / g
    float*  ybuf = (float*) alloc((size_t)NPATCH * PPX * 64 * 4);  // y (patch layout)

    auto blk = [](long long e) { return (unsigned)((e + 255) / 256); };

    // 0) msum = 0, aggregate per-patch expert weights (f16) + biases (f32)
    zero_msum_kernel<<<1, 128, 0, stream>>>(msum);
    agg_pw_kernel<<<blk((long long)NPATCH * 128 * 65), 256, 0, stream>>>(w1, b1, v, aw1h, ab1, 128, 64);
    agg_w2_kernel<<<blk((long long)NPATCH * (9 * 128 * 128 + 128)), 256, 0, stream>>>(w2, b2, v, aw2h, ab2);
    agg_pw_kernel<<<blk((long long)NPATCH * 64 * 65), 256, 0, stream>>>(wsca, bsca, v, awsh, absc, 64, 64);
    agg_pw_kernel<<<blk((long long)NPATCH * 64 * 65), 256, 0, stream>>>(w3, b3, v, aw3h, ab3, 64, 64);
    agg_pw_kernel<<<blk((long long)NPATCH * 128 * 65), 256, 0, stream>>>(w4, b4, v, aw4h, ab4, 128, 64);
    agg_pw_kernel<<<blk((long long)NPATCH * 64 * 65), 256, 0, stream>>>(w5, b5, v, aw5h, ab5, 64, 64);

    // 1) LN1: NCHW -> patch layout f16
    ln1_kernel<<<blk(NPIX), 256, 0, stream>>>(inp, n1w, n1b, bufC);

    // 2) conv1 (64->128): x1 = aw1 * xln + b1
    gemm_pw_kernel<128, 64, 0><<<4608, 256, 0, stream>>>(aw1h, ab1, bufC, bufA,
                                                         nullptr, nullptr, nullptr, nullptr);
    // 3) conv2 3x3 (128->128), LDS-staged
    conv3x3_kernel<<<NPATCH * 64, 256, 0, stream>>>(aw2h, ab2, bufA, bufB);

    // 4) SimpleGate + global spatial mean
    gate_kernel<<<NPIX / 64, 256, 0, stream>>>(bufB, bufC, msum, 1);

    // 5) SCA: t = mean * (awsca * keep + bsca)
    gemm_pw_kernel<64, 64, 1><<<2304, 256, 0, stream>>>(awsh, absc, bufC, bufD,
                                                        nullptr, msum, nullptr, nullptr);
    // 6) conv3 (64->64) + residual: y = inp + o*beta  (f32 patch layout)
    gemm_pw_kernel<64, 64, 2><<<2304, 256, 0, stream>>>(aw3h, ab3, bufD, nullptr,
                                                        ybuf, nullptr, inp, beta);
    // 7) LN2 on y
    ln2_kernel<<<blk((long long)NPATCH * PPX), 256, 0, stream>>>(ybuf, n2w, n2b, bufC);

    // 8) conv4 (64->128)
    gemm_pw_kernel<128, 64, 0><<<4608, 256, 0, stream>>>(aw4h, ab4, bufC, bufA,
                                                         nullptr, nullptr, nullptr, nullptr);
    // 9) SimpleGate (no mean)
    gate_kernel<<<NPIX / 64, 256, 0, stream>>>(bufA, bufD, nullptr, 0);

    // 10) conv5 (64->64) + out = y + o*gamma  -> NCHW f32 d_out
    gemm_pw_kernel<64, 64, 3><<<2304, 256, 0, stream>>>(aw5h, ab5, bufD, nullptr,
                                                        dout, nullptr, ybuf, gam);

    // 11) second tuple output: v
    (void)hipMemcpyAsync(dout + (size_t)2 * 64 * 384 * 384, d_in[1], 360 * sizeof(float),
                         hipMemcpyDeviceToDevice, stream);
}
